// RNNModel_90546500534313
// MI455X (gfx1250) — compile-verified, matrix operand order checked by reading, not run on hardware
//
#include <hip/hip_runtime.h>
#include <hip/hip_fp16.h>

// MoS-LSTM LM for MI455X (gfx1250, wave32, WMMA).
// Head GEMM (5120x30000,K=280) dominates: f16 WMMA, f32 accum, two-pass online
// softmax so the 614MB logit tensor never exists. dec_W f16 (17MB) lives in L2;
// cooperative LDS staging of dec tiles (8 waves/block share one tile stream,
// double-buffered, async global->LDS prefetch) cuts L2 read traffic 8x
// (~8.8GB -> ~1.1GB per pass), which is the binding roofline.

typedef __attribute__((ext_vector_type(16))) _Float16 v16h;
typedef __attribute__((ext_vector_type(8)))  _Float16 v8h;
typedef __attribute__((ext_vector_type(8)))  float    v8f;
typedef __attribute__((ext_vector_type(4)))  int      v4i;

#define WMMA(a, b, c) __builtin_amdgcn_wmma_f32_16x16x32_f16( \
    false, (a), false, (b), (short)0, (c), false, false)

#if defined(__has_builtin)
#if __has_builtin(__builtin_amdgcn_global_load_async_to_lds_b128)
#define HAVE_ASYNC_LDS 1
#endif
#endif
#ifndef HAVE_ASYNC_LDS
#define HAVE_ASYNC_LDS 0
#endif

#if HAVE_ASYNC_LDS
typedef __attribute__((address_space(1))) v4i gv4i;  // global (AS1, 64-bit)
typedef __attribute__((address_space(3))) v4i lv4i;  // LDS (AS3, 32-bit)
#endif

__device__ __forceinline__ void async_wait0() {
#if HAVE_ASYNC_LDS
  asm volatile("s_wait_asynccnt 0x0" ::: "memory");
#endif
}

// Stage one 16x288 f16 tile (9216B = 576 x b128 chunks) into LDS, 256 threads.
__device__ __forceinline__ void stage_tile(const _Float16* __restrict__ dec,
                                           int nt, _Float16* __restrict__ buf,
                                           int tid) {
  for (int c = tid; c < 576; c += 256) {
    const _Float16* g = dec + (long)(nt * 16 + c / 36) * 288 + (c % 36) * 8;
    _Float16* l = buf + c * 8;
#if HAVE_ASYNC_LDS
    // AS3 pointer = low 32 bits of the flat LDS-aperture address (ISA 10.2).
    __builtin_amdgcn_global_load_async_to_lds_b128(
        (gv4i*)(uintptr_t)g, (lv4i*)(unsigned int)(uintptr_t)l, 0, 0);
#else
    *(v8h*)l = *(const v8h*)g;
#endif
  }
}

// CDNA5 16-bit A/B fragment layout: lane L -> row (L&15); lane-half hf=L>>4:
// elements 0..7 = row[k0+8*hf .. +8), elements 8..15 = row[k0+16+8*hf .. +8).
__device__ __forceinline__ v16h load_frag(const _Float16* row, int k0, int hf) {
  union { v16h v; v8h h[2]; } u;
  u.h[0] = *(const v8h*)(row + k0 + 8 * hf);
  u.h[1] = *(const v8h*)(row + k0 + 16 + 8 * hf);
  return u.v;
}

// ---------------- utility kernels ----------------

__global__ void embed_kernel(const int* __restrict__ idx,
                             const float* __restrict__ embW,
                             float* __restrict__ x0, int total) {
  int i = blockIdx.x * blockDim.x + threadIdx.x;
  if (i >= total) return;
  int r = i / 280, k = i % 280;
  x0[i] = embW[(long)idx[r] * 280 + k];
}

// f32 (ng*hid rows x K) -> f16 (ng*hidp rows x Kp), zero pad rows/cols,
// per-gate padding so gate q starts at row q*hidp.
__global__ void cvt_w(const float* __restrict__ src, _Float16* __restrict__ dst,
                      int hid, int hidp, int K, int Kp, int total) {
  int i = blockIdx.x * blockDim.x + threadIdx.x;
  if (i >= total) return;
  int kp = i % Kp, row = i / Kp;
  int q = row / hidp, u = row % hidp;
  float v = (u < hid && kp < K) ? src[((long)(q * hid + u)) * K + kp] : 0.f;
  dst[i] = (_Float16)v;
}

__global__ void bias_pad(const float* __restrict__ bih, const float* __restrict__ bhh,
                         float* __restrict__ bp, int hid, int hidp, int total) {
  int i = blockIdx.x * blockDim.x + threadIdx.x;
  if (i >= total) return;
  int q = i / hidp, u = i % hidp;
  bp[i] = (u < hid) ? bih[q * hid + u] + bhh[q * hid + u] : 0.f;
}

__global__ void zero_f16(_Float16* __restrict__ p, int n) {
  int i = blockIdx.x * blockDim.x + threadIdx.x;
  if (i < n) p[i] = (_Float16)0.f;
}

// init per-layer state: h double buffers (16 x Kp, f16) and c (8 x hidp, f32)
__global__ void init_layer(const float* __restrict__ h0, const float* __restrict__ c0,
                           _Float16* __restrict__ hb0, _Float16* __restrict__ hb1,
                           float* __restrict__ c, int hid, int hidp, int Kp) {
  int i = blockIdx.x * blockDim.x + threadIdx.x;
  if (i < 16 * Kp) {
    int m = i / Kp, n = i % Kp;
    _Float16 v = (m < 8 && n < hid) ? (_Float16)h0[m * hid + n] : (_Float16)0.f;
    hb0[i] = v;
    hb1[i] = (_Float16)0.f;
  }
  if (i < 8 * hidp) {
    int m = i / hidp, n = i % hidp;
    c[i] = (n < hid) ? c0[m * hid + n] : 0.f;
  }
}

// ---------------- WMMA GEMM: C[M,N] = A(MxKp) * B(NxKp)^T (+bias, epilogue) ----
// MODE 0: f32 out to Cout (ld=N), + bias. MODE 1: tanh, f16 scatter into
// padded ctx layout: column n -> (e=n/280, k=n%280), row m -> ctx[(m*16+e)*288+k].
template <int MODE>
__global__ __launch_bounds__(128) void gemm_wmma(
    const _Float16* __restrict__ A, const _Float16* __restrict__ Bw,
    const float* __restrict__ bias, float* __restrict__ Cout,
    _Float16* __restrict__ ctx, int N, int Kp) {
  int mtile = blockIdx.x;
  int ntile = blockIdx.y * (blockDim.x >> 5) + (threadIdx.x >> 5);
  if (ntile * 16 >= N) return;  // wave-uniform
  int lane = threadIdx.x & 31, hf = lane >> 4, lm = lane & 15;
  const _Float16* arow = A + (long)(mtile * 16 + lm) * Kp;
  const _Float16* brow = Bw + (long)(ntile * 16 + lm) * Kp;
  v8f acc = {};
  for (int k0 = 0; k0 < Kp; k0 += 32) {
    v16h a = load_frag(arow, k0, hf);
    v16h b = load_frag(brow, k0, hf);
    acc = WMMA(a, b, acc);
  }
  int n = ntile * 16 + lm;
  float bv = bias ? bias[n] : 0.f;
#pragma unroll
  for (int j = 0; j < 8; ++j) {
    int m = mtile * 16 + j + 8 * hf;
    float val = acc[j] + bv;
    if (MODE == 0) {
      Cout[(long)m * N + n] = val;
    } else {
      int e = n / 280, k = n % 280;
      ctx[((long)m * 16 + e) * 288 + k] = (_Float16)tanhf(val);
    }
  }
}

// ---------------- fused LSTM step ----------------
// g = gx[t] + h_prev @ Whh^T (4 gate tiles per wave), then gates + c/h update.
__global__ __launch_bounds__(128) void lstm_step(
    const _Float16* __restrict__ hprev, const _Float16* __restrict__ Whh,
    const float* __restrict__ gx, float* __restrict__ c,
    float* __restrict__ xout, _Float16* __restrict__ hnext,
    int t, int hid, int hidp, int Kp) {
  int ntile = blockIdx.x * (blockDim.x >> 5) + (threadIdx.x >> 5);
  if (ntile * 16 >= hid) return;  // wave-uniform
  int lane = threadIdx.x & 31, hf = lane >> 4, lm = lane & 15;
  const _Float16* arow = hprev + (long)lm * Kp;
  v8f acc[4] = {};
  for (int k0 = 0; k0 < Kp; k0 += 32) {
    v16h a = load_frag(arow, k0, hf);
#pragma unroll
    for (int q = 0; q < 4; ++q) {
      const _Float16* brow = Whh + (long)(q * hidp + ntile * 16 + lm) * Kp;
      v16h b = load_frag(brow, k0, hf);
      acc[q] = WMMA(a, b, acc[q]);
    }
  }
  int n = ntile * 16 + lm;
  int NG = 4 * hidp;
  if (n < hid) {
    const float* gxr = gx + (long)(t * 8) * NG;
#pragma unroll
    for (int j = 0; j < 8; ++j) {
      int m = j + 8 * hf;
      if (m < 8) {  // rows 8..15 are batch padding
        const float* g = gxr + (long)m * NG;
        float gi = acc[0][j] + g[0 * hidp + n];
        float gf = acc[1][j] + g[1 * hidp + n];
        float gg = acc[2][j] + g[2 * hidp + n];
        float go = acc[3][j] + g[3 * hidp + n];
        float si = 1.f / (1.f + __expf(-gi));
        float sf = 1.f / (1.f + __expf(-gf));
        float so = 1.f / (1.f + __expf(-go));
        float cc = sf * c[m * hidp + n] + si * tanhf(gg);
        c[m * hidp + n] = cc;
        float hh = so * tanhf(cc);
        xout[(long)(t * 8 + m) * hid + n] = hh;
        hnext[(long)m * Kp + n] = (_Float16)hh;
      }
    }
  }
}

// ---------------- prior softmax: (512 x 10) over K=620 ----------------
__global__ void prior_kernel(const float* __restrict__ x3,
                             const float* __restrict__ pw,
                             float* __restrict__ prior) {
  int tb = blockIdx.x;
  int e = threadIdx.x >> 5, lane = threadIdx.x & 31;
  const float* xr = x3 + (long)tb * 620;
  const float* wr = pw + (long)e * 620;
  float s = 0.f;
  for (int k = lane; k < 620; k += 32) s += xr[k] * wr[k];
  for (int d = 16; d; d >>= 1) s += __shfl_xor(s, d, 32);
  __shared__ float sh[10];
  if (lane == 0) sh[e] = s;
  __syncthreads();
  if (threadIdx.x == 0) {
    float m = -__builtin_inff();
    for (int i = 0; i < 10; ++i) m = fmaxf(m, sh[i]);
    float z = 0.f, ex[10];
    for (int i = 0; i < 10; ++i) { ex[i] = __expf(sh[i] - m); z += ex[i]; }
    for (int i = 0; i < 10; ++i) prior[tb * 10 + i] = ex[i] / z;
  }
}

// ---------------- head pass A: per-row (tb,e) running max / sum-exp -----------
// Block = 8 waves = 8 tb rows sharing one dec-tile stream staged in LDS
// (double-buffered, async global->LDS prefetch of tile n+1 during tile n).
#define NSTRIP 30
__global__ __launch_bounds__(256) void head_passA(
    const _Float16* __restrict__ ctx, const _Float16* __restrict__ dec,
    const float* __restrict__ dec_b, float2* __restrict__ part) {
  __shared__ _Float16 bt[2][16 * 288];
  int tid = threadIdx.x;
  int wave = tid >> 5, lane = tid & 31, hf = lane >> 4, lm = lane & 15;
  int tb = blockIdx.x * 8 + wave;
  int strip = blockIdx.y;
  const _Float16* arow = ctx + (long)(tb * 16 + lm) * 288;
  v16h afr[9];
#pragma unroll
  for (int kk = 0; kk < 9; ++kk) afr[kk] = load_frag(arow, kk * 32, hf);
  float rm[8], rs[8];
#pragma unroll
  for (int j = 0; j < 8; ++j) { rm[j] = -__builtin_inff(); rs[j] = 0.f; }

  stage_tile(dec, strip, bt[0], tid);
  async_wait0();
  __syncthreads();
  int p = 0;
  for (int nt = strip; nt < 1875; nt += NSTRIP) {
    int ntn = nt + NSTRIP;
    if (ntn < 1875) stage_tile(dec, ntn, bt[p ^ 1], tid);  // prefetch
    const _Float16* brow = &bt[p][lm * 288];
    v8f acc = {};
#pragma unroll
    for (int kk = 0; kk < 9; ++kk) {
      v16h b = load_frag(brow, kk * 32, hf);  // ds_load_b128 x2
      acc = WMMA(afr[kk], b, acc);
    }
    float db = dec_b[nt * 16 + lm];
#pragma unroll
    for (int j = 0; j < 8; ++j) {
      float x = acc[j] + db;
      float nm = fmaxf(rm[j], x);
      rs[j] = rs[j] * __expf(rm[j] - nm) + __expf(x - nm);
      rm[j] = nm;
    }
    async_wait0();
    __syncthreads();
    p ^= 1;
  }
  // row-wise combine across the 16 lanes of each half (same rows, diff cols)
  for (int d = 1; d < 16; d <<= 1) {
#pragma unroll
    for (int j = 0; j < 8; ++j) {
      float om = __shfl_xor(rm[j], d, 32);
      float os = __shfl_xor(rs[j], d, 32);
      float nm = fmaxf(rm[j], om);
      rs[j] = rs[j] * __expf(rm[j] - nm) + os * __expf(om - nm);
      rm[j] = nm;
    }
  }
  if (lm == 0) {
#pragma unroll
    for (int j = 0; j < 8; ++j) {
      int e = j + 8 * hf;
      part[(long)(tb * 16 + e) * NSTRIP + strip] = make_float2(rm[j], rs[j]);
    }
  }
}

__global__ void head_reduce(const float2* __restrict__ part,
                            const float* __restrict__ prior,
                            float* __restrict__ coef, float* __restrict__ rowmax) {
  int r = blockIdx.x * blockDim.x + threadIdx.x;
  if (r >= 512 * 16) return;
  float M = -__builtin_inff(), Z = 0.f;
  for (int i = 0; i < NSTRIP; ++i) {
    float2 p = part[(long)r * NSTRIP + i];
    float nm = fmaxf(M, p.x);
    Z = Z * __expf(M - nm) + p.y * __expf(p.x - nm);
    M = nm;
  }
  int e = r & 15, tb = r >> 4;
  rowmax[r] = M;
  coef[r] = (e < 10) ? prior[tb * 10 + e] / Z : 0.f;
}

// ---------------- head pass B: recompute logits, mixture, log ----------------
__global__ __launch_bounds__(256) void head_passB(
    const _Float16* __restrict__ ctx, const _Float16* __restrict__ dec,
    const float* __restrict__ dec_b, const float* __restrict__ coef,
    const float* __restrict__ rowmax, float* __restrict__ out) {
  __shared__ _Float16 bt[2][16 * 288];
  int tid = threadIdx.x;
  int wave = tid >> 5, lane = tid & 31, hf = lane >> 4, lm = lane & 15;
  int tb = blockIdx.x * 8 + wave;
  int strip = blockIdx.y;
  const _Float16* arow = ctx + (long)(tb * 16 + lm) * 288;
  v16h afr[9];
#pragma unroll
  for (int kk = 0; kk < 9; ++kk) afr[kk] = load_frag(arow, kk * 32, hf);
  float cf[8], rmx[8];
#pragma unroll
  for (int j = 0; j < 8; ++j) {
    int e = j + 8 * hf;
    cf[j] = coef[tb * 16 + e];
    rmx[j] = rowmax[tb * 16 + e];
  }
  stage_tile(dec, strip, bt[0], tid);
  async_wait0();
  __syncthreads();
  int p = 0;
  for (int nt = strip; nt < 1875; nt += NSTRIP) {
    int ntn = nt + NSTRIP;
    if (ntn < 1875) stage_tile(dec, ntn, bt[p ^ 1], tid);  // prefetch
    const _Float16* brow = &bt[p][lm * 288];
    v8f acc = {};
#pragma unroll
    for (int kk = 0; kk < 9; ++kk) {
      v16h b = load_frag(brow, kk * 32, hf);
      acc = WMMA(afr[kk], b, acc);
    }
    float db = dec_b[nt * 16 + lm];
    float ps = 0.f;
#pragma unroll
    for (int j = 0; j < 8; ++j)
      ps += cf[j] * __expf(acc[j] + db - rmx[j]);
    ps += __shfl_xor(ps, 16, 32);  // add experts 8..15 half
    if (lane < 16)
      out[(long)tb * 30000 + nt * 16 + lane] = logf(ps + 1e-8f);
    async_wait0();
    __syncthreads();
    p ^= 1;
  }
}

// ---------------- host orchestration ----------------

static inline int ceil_div(int a, int b) { return (a + b - 1) / b; }

extern "C" void kernel_launch(void* const* d_in, const int* in_sizes, int n_in,
                              void* d_out, int out_size, void* d_ws, size_t ws_size,
                              hipStream_t stream) {
  (void)in_sizes; (void)n_in; (void)out_size; (void)ws_size;
  const int TB = 512;
  const int NTOK = 30000;

  const int*   input  = (const int*)d_in[0];
  const float* h0[3]  = {(const float*)d_in[1], (const float*)d_in[3], (const float*)d_in[5]};
  const float* c0[3]  = {(const float*)d_in[2], (const float*)d_in[4], (const float*)d_in[6]};
  const float* embW   = (const float*)d_in[7];
  const float *Wih[3], *Whh[3], *bih[3], *bhh[3];
  for (int l = 0; l < 3; ++l) {
    Wih[l] = (const float*)d_in[8 + 4 * l];
    Whh[l] = (const float*)d_in[9 + 4 * l];
    bih[l] = (const float*)d_in[10 + 4 * l];
    bhh[l] = (const float*)d_in[11 + 4 * l];
  }
  const float* latent_W = (const float*)d_in[20];
  const float* latent_b = (const float*)d_in[21];
  const float* prior_W  = (const float*)d_in[22];
  const float* dec_W    = (const float*)d_in[23];
  const float* dec_b    = (const float*)d_in[24];
  float* out = (float*)d_out;

  const int in_dim[3] = {280, 960, 960};
  const int kin_p[3]  = {288, 960, 960};
  const int hid[3]    = {960, 960, 620};
  const int hidp[3]   = {960, 960, 640};

  // workspace carve-up (256B aligned)
  char* base = (char*)d_ws;
  size_t off = 0;
  auto alloc = [&](size_t bytes) -> void* {
    void* r = base + off;
    off = (off + bytes + 255) & ~(size_t)255;
    return r;
  };
  float* x0 = (float*)alloc((size_t)TB * 280 * 4);
  float* x1 = (float*)alloc((size_t)TB * 960 * 4);
  float* x2 = (float*)alloc((size_t)TB * 960 * 4);
  float* x3 = (float*)alloc((size_t)TB * 620 * 4);
  float* xb[4] = {x0, x1, x2, x3};
  _Float16* xf   = (_Float16*)alloc((size_t)TB * 960 * 2);
  float*    gx   = (float*)alloc((size_t)TB * 3840 * 4);
  _Float16* wA   = (_Float16*)alloc((size_t)3840 * 960 * 2);
  _Float16* wB   = (_Float16*)alloc((size_t)3840 * 960 * 2);
  float*    bp   = (float*)alloc((size_t)3840 * 4);
  _Float16* hb0  = (_Float16*)alloc((size_t)16 * 960 * 2);
  _Float16* hb1  = (_Float16*)alloc((size_t)16 * 960 * 2);
  float*    cbuf = (float*)alloc((size_t)8 * 960 * 4);
  _Float16* outf = (_Float16*)alloc((size_t)TB * 640 * 2);
  _Float16* latf = (_Float16*)alloc((size_t)2800 * 640 * 2);
  _Float16* ctx  = (_Float16*)alloc((size_t)TB * 16 * 288 * 2);
  _Float16* decf = (_Float16*)alloc((size_t)NTOK * 288 * 2);
  float* prior   = (float*)alloc((size_t)TB * 10 * 4);
  float* coef    = (float*)alloc((size_t)8192 * 4);
  float* rowmax  = (float*)alloc((size_t)8192 * 4);
  float2* partA  = (float2*)gx;  // LSTM done before head pass A; reuse

  // 1) embedding gather
  {
    int total = TB * 280;
    embed_kernel<<<ceil_div(total, 256), 256, 0, stream>>>(input, embW, x0, total);
  }
  // 2) one-time weight conversions for the head
  {
    int total = NTOK * 288;
    cvt_w<<<ceil_div(total, 256), 256, 0, stream>>>(dec_W, decf, NTOK, NTOK, 280, 288, total);
    total = 2800 * 640;
    cvt_w<<<ceil_div(total, 256), 256, 0, stream>>>(latent_W, latf, 2800, 2800, 620, 640, total);
    total = TB * 16 * 288;
    zero_f16<<<ceil_div(total, 256), 256, 0, stream>>>(ctx, total);
  }
  // 3) LSTM layers
  for (int l = 0; l < 3; ++l) {
    int K = in_dim[l], Kp = kin_p[l], H = hid[l], Hp = hidp[l];
    int NG = 4 * Hp;
    {  // x -> f16 padded
      int total = TB * Kp;
      cvt_w<<<ceil_div(total, 256), 256, 0, stream>>>(xb[l], xf, TB, TB, K, Kp, total);
    }
    {  // Wih -> f16, per-gate padded rows
      int total = NG * Kp;
      cvt_w<<<ceil_div(total, 256), 256, 0, stream>>>(Wih[l], wA, H, Hp, K, Kp, total);
    }
    {  // bias
      int total = NG;
      bias_pad<<<ceil_div(total, 256), 256, 0, stream>>>(bih[l], bhh[l], bp, H, Hp, total);
    }
    {  // gx = x @ Wih^T + b  (M=512, N=4*Hp, K=Kp)
      int ntiles = NG / 16;
      dim3 grid(TB / 16, ceil_div(ntiles, 4));
      gemm_wmma<0><<<grid, 128, 0, stream>>>(xf, wA, bp, gx, nullptr, NG, Kp);
    }
    {  // Whh -> f16 padded (K dim = H padded to Hp)
      int total = NG * Hp;
      cvt_w<<<ceil_div(total, 256), 256, 0, stream>>>(Whh[l], wB, H, Hp, H, Hp, total);
    }
    {  // state init
      int total = 16 * Hp;
      init_layer<<<ceil_div(total, 256), 256, 0, stream>>>(h0[l], c0[l], hb0, hb1, cbuf, H, Hp, Hp);
    }
    int ntiles = ceil_div(H, 16);
    dim3 sgrid(ceil_div(ntiles, 4));
    for (int t = 0; t < 64; ++t) {
      _Float16* hp = (t & 1) ? hb1 : hb0;
      _Float16* hn = (t & 1) ? hb0 : hb1;
      lstm_step<<<sgrid, 128, 0, stream>>>(hp, wB, gx, cbuf, xb[l + 1], hn, t, H, Hp, Hp);
    }
  }
  // 4) head
  {  // out(f32 620) -> f16 640
    int total = TB * 640;
    cvt_w<<<ceil_div(total, 256), 256, 0, stream>>>(x3, outf, TB, TB, 620, 640, total);
  }
  {  // latent = tanh(out @ latent_W^T + b) scattered into padded ctx
    dim3 grid(TB / 16, ceil_div(2800 / 16, 4));
    gemm_wmma<1><<<grid, 128, 0, stream>>>(outf, latf, latent_b, nullptr, ctx, 2800, 640);
  }
  prior_kernel<<<TB, 320, 0, stream>>>(x3, prior_W, prior);
  {
    dim3 grid(TB / 8, NSTRIP);
    head_passA<<<grid, 256, 0, stream>>>(ctx, decf, dec_b, partA);
  }
  head_reduce<<<ceil_div(8192, 256), 256, 0, stream>>>(partA, prior, coef, rowmax);
  {
    dim3 grid(TB / 8, NSTRIP);
    head_passB<<<grid, 256, 0, stream>>>(ctx, decf, dec_b, coef, rowmax, out);
  }
}
